// MoELayer_78254304133257
// MI455X (gfx1250) — matrix operand order repaired and checked by reading, not compiled, not run on hardware
//
#include <hip/hip_runtime.h>

// ---------------- problem constants ----------------
#define D_    1024
#define H_    4096
#define E_    8
#define T_    4096        // 2*2048 tokens
#define KTOP  2
#define HC    128         // h-chunk processed per iteration
#define NCHUNK (H_ / HC)  // 32
#define MAX_TILES ((T_*KTOP + E_*15 + 15) / 16)   // 520 worst-case padded tiles

typedef __attribute__((ext_vector_type(16))) __bf16 v16bf;
typedef __attribute__((ext_vector_type(8)))  __bf16 v8bf;
typedef __attribute__((ext_vector_type(8)))  float  v8f;

// float -> bf16 via native fptrunc (lowers to v_cvt_pk_bf16_f32 on gfx1250)
static __device__ __forceinline__ __bf16 f2bf(float f) { return (__bf16)f; }

// A-fragment for v_wmma_f32_16x16x32_bf16 (16x32 bf16; lane m = lane%16,
// hi = lane/16). Per ISA 16-bit A layout: elems 0..7 <-> k = 8*hi+[0..8),
// elems 8..15 <-> k = 16+8*hi+[0..8). Two 32B vector loads + packed cvts.
static __device__ __forceinline__ v16bf load_a_frag_f32(const float* __restrict__ row, int hi) {
    v8f f0 = *reinterpret_cast<const v8f*>(row + 8 * hi);
    v8f f1 = *reinterpret_cast<const v8f*>(row + 16 + 8 * hi);
    v8bf b0 = __builtin_convertvector(f0, v8bf);
    v8bf b1 = __builtin_convertvector(f1, v8bf);
    return __builtin_shufflevector(b0, b1, 0, 1, 2, 3, 4, 5, 6, 7,
                                           8, 9, 10, 11, 12, 13, 14, 15);
}

// ---------------- kernel 1: router (wave per token) ----------------
__global__ __launch_bounds__(256) void moe_router(
    const float* __restrict__ x, const float* __restrict__ Wr, const float* __restrict__ br,
    int* __restrict__ counts, int* __restrict__ a_e, int* __restrict__ a_slot,
    float* __restrict__ a_g)
{
    int lane = threadIdx.x & 31;
    int wave = threadIdx.x >> 5;
    int t = blockIdx.x * 8 + wave;
    if (t >= T_) return;
    const float* xr = x + (size_t)t * D_;

    float logit[E_];
    for (int e = 0; e < E_; ++e) {
        const float* wr = Wr + (size_t)e * D_;
        float s = 0.f;
        for (int d = lane; d < D_; d += 32) s += xr[d] * wr[d];
        for (int off = 16; off; off >>= 1) s += __shfl_xor(s, off);
        logit[e] = s + br[e];
    }
    if (lane == 0) {
        float mx = logit[0];
        for (int e = 1; e < E_; ++e) mx = fmaxf(mx, logit[e]);
        float g[E_], sum = 0.f;
        for (int e = 0; e < E_; ++e) { g[e] = expf(logit[e] - mx); sum += g[e]; }
        float inv = 1.f / sum;
        for (int e = 0; e < E_; ++e) g[e] *= inv;
        // top-2 (first-max tie break, matching lax.top_k)
        int i0 = 0;
        for (int e = 1; e < E_; ++e) if (g[e] > g[i0]) i0 = e;
        int i1 = -1;
        for (int e = 0; e < E_; ++e) {
            if (e == i0) continue;
            if (i1 < 0 || g[e] > g[i1]) i1 = e;
        }
        int s0 = atomicAdd(&counts[i0], 1);
        int s1 = atomicAdd(&counts[i1], 1);
        a_e[t * 2 + 0] = i0; a_slot[t * 2 + 0] = s0; a_g[t * 2 + 0] = g[i0];
        a_e[t * 2 + 1] = i1; a_slot[t * 2 + 1] = s1; a_g[t * 2 + 1] = g[i1];
    }
}

// ---------------- kernel 2: padded prefix + tile map ----------------
__global__ void moe_prefix(const int* __restrict__ counts, int* __restrict__ pbase,
                           int* __restrict__ tileExpert, int* __restrict__ perm,
                           float* __restrict__ gslot)
{
    if (threadIdx.x == 0 && blockIdx.x == 0) {
        int base = 0, tile = 0;
        for (int e = 0; e < E_; ++e) {
            pbase[e] = base;
            int c = counts[e];
            int padded = (c + 15) & ~15;
            for (int s = c; s < padded; ++s) { perm[base + s] = 0; gslot[base + s] = 0.f; }
            for (int i = 0; i < (padded >> 4); ++i) tileExpert[tile++] = e;
            base += padded;
        }
        pbase[E_] = base;
        for (; tile < MAX_TILES; ++tile) tileExpert[tile] = -1;
    }
}

// ---------------- kernel 3: scatter slot -> token ----------------
__global__ __launch_bounds__(256) void moe_scatter(
    const int* __restrict__ a_e, const int* __restrict__ a_slot, const float* __restrict__ a_g,
    const int* __restrict__ pbase, int* __restrict__ perm, float* __restrict__ gslot)
{
    int i = blockIdx.x * blockDim.x + threadIdx.x;
    if (i >= T_ * KTOP) return;
    int e  = a_e[i];
    int gs = pbase[e] + a_slot[i];
    perm[gs]  = i >> 1;      // token id
    gslot[gs] = a_g[i];
}

// ---------------- kernel 4: fused expert MLP (WMMA bf16) ----------------
// Block = one tile of 16 slots of a single expert; 8 waves.
// GEMM1: h^T[16hcols x 16tok] per wave (K=D). Fragments are loaded in groups
//        of 4 into disjoint registers, then 4 hazard-free WMMAs issue
//        back-to-back into 4 interleaved accumulators.
// GEMM2: each wave owns 128 output dims; B fragments hoisted per chunk,
//        A fragments grouped 4-wide, 8 independent accumulators.
__global__ __launch_bounds__(256) void moe_mlp(
    const float* __restrict__ x,
    const float* __restrict__ W1, const float* __restrict__ b1,
    const float* __restrict__ W2, const float* __restrict__ b2,
    const int* __restrict__ perm, const float* __restrict__ gslot,
    const int* __restrict__ tileExpert, float* __restrict__ out)
{
    __shared__ __attribute__((aligned(32))) __bf16 xsT[D_ * 16];  // x tile [16 tok][D], 32 KB
    __shared__ __attribute__((aligned(32))) __bf16 hsT[HC * 16];  // h^T chunk, 4 KB
    __shared__ float gS[16];
    __shared__ int   tS[16];

    int e = tileExpert[blockIdx.x];
    if (e < 0) return;
    int tid = threadIdx.x;

    if (tid < 16) {
        int s = blockIdx.x * 16 + tid;
        tS[tid] = perm[s];
        gS[tid] = gslot[s];
    }
    __syncthreads();

    // stage x^T (bf16) : 16 threads per token row, coalesced 64B bursts
    {
        int n = tid >> 4;
        int c = tid & 15;
        const float* xr = x + (size_t)tS[n] * D_;
        for (int j = 0; j < D_ / 16; ++j) {
            int d = c + j * 16;
            xsT[n * D_ + d] = f2bf(xr[d]);
        }
    }
    __syncthreads();

    int lane = tid & 31;
    int wave = tid >> 5;
    int n    = lane & 15;   // token column (B/C layout: lanes <-> N)
    int hi   = lane >> 4;
    float gn = gS[n];
    int tokn = tS[n];

    const float* W1e = W1 + (size_t)e * H_ * D_;
    const float* W2e = W2 + (size_t)e * D_ * H_;
    const float* b1e = b1 + (size_t)e * H_;
    const float* b2e = b2 + (size_t)e * D_;

    v8f acc[8];
    for (int i = 0; i < 8; ++i) acc[i] = (v8f){0.f, 0.f, 0.f, 0.f, 0.f, 0.f, 0.f, 0.f};

    for (int hc = 0; hc < NCHUNK; ++hc) {
        // -------- GEMM1: wave computes hcols [hc*HC + wave*16, +16) --------
        v8f c4[4];
        for (int i = 0; i < 4; ++i) c4[i] = (v8f){0.f, 0.f, 0.f, 0.f, 0.f, 0.f, 0.f, 0.f};
        const float* arow = W1e + (size_t)(hc * HC + wave * 16 + n) * D_;

        #pragma unroll 2
        for (int kg = 0; kg < D_ / 128; ++kg) {        // 8 groups of 4 k-steps
            int kb0 = kg * 128;
            __builtin_prefetch(arow + kb0 + 512, 0, 0);
            v16bf a4[4], b4[4];
            #pragma unroll
            for (int i = 0; i < 4; ++i) {              // batched load+convert phase
                int kb = kb0 + i * 32;
                a4[i] = load_a_frag_f32(arow + kb, hi);
                v8bf bl = *reinterpret_cast<const v8bf*>(&xsT[n * D_ + kb + 8 * hi]);
                v8bf bh = *reinterpret_cast<const v8bf*>(&xsT[n * D_ + kb + 16 + 8 * hi]);
                b4[i] = __builtin_shufflevector(bl, bh, 0, 1, 2, 3, 4, 5, 6, 7,
                                                8, 9, 10, 11, 12, 13, 14, 15);
            }
            #pragma unroll
            for (int i = 0; i < 4; ++i)                // 4 hazard-free WMMAs
                c4[i] = __builtin_amdgcn_wmma_f32_16x16x32_bf16(false, a4[i], false, b4[i],
                                                                (short)0, c4[i],
                                                                false, false);
        }
        v8f c1 = (c4[0] + c4[1]) + (c4[2] + c4[3]);

        // epilogue: bias + exact-erf GELU + gate, write h^T chunk as bf16
        #pragma unroll
        for (int r = 0; r < 8; ++r) {
            int m    = r + 8 * hi;               // hcol within wave's 16
            int hcol = hc * HC + wave * 16 + m;
            float v  = c1[r] + b1e[hcol];
            v = 0.5f * v * (1.f + erff(v * 0.70710678118654752f));
            hsT[n * HC + wave * 16 + m] = f2bf(v * gn);
        }
        __syncthreads();   // h^T chunk visible to all waves

        // -------- GEMM2: wave owns dcols [wave*128, +128), K = this chunk ------
        // Hoist the 4 B fragments once (they depend only on ks, not mt).
        v16bf bfr[4];
        #pragma unroll
        for (int ks = 0; ks < HC / 32; ++ks) {
            v8bf bl = *reinterpret_cast<const v8bf*>(&hsT[n * HC + ks * 32 + 8 * hi]);
            v8bf bh = *reinterpret_cast<const v8bf*>(&hsT[n * HC + ks * 32 + 16 + 8 * hi]);
            bfr[ks] = __builtin_shufflevector(bl, bh, 0, 1, 2, 3, 4, 5, 6, 7,
                                              8, 9, 10, 11, 12, 13, 14, 15);
        }

        // Prefetch next chunk's W2 columns as one burst.
        if (hc + 1 < NCHUNK) {
            #pragma unroll
            for (int mt = 0; mt < 8; ++mt)
                __builtin_prefetch(W2e + (size_t)(wave * 128 + mt * 16 + n) * H_
                                       + (hc + 1) * HC, 0, 0);
        }

        #pragma unroll
        for (int ks = 0; ks < HC / 32; ++ks) {
            #pragma unroll
            for (int half = 0; half < 2; ++half) {
                v16bf a4[4];
                #pragma unroll
                for (int i = 0; i < 4; ++i) {          // batched load+convert phase
                    int mt = half * 4 + i;
                    a4[i] = load_a_frag_f32(W2e + (size_t)(wave * 128 + mt * 16 + n) * H_
                                                + hc * HC + ks * 32, hi);
                }
                #pragma unroll
                for (int i = 0; i < 4; ++i) {          // 4 hazard-free WMMAs
                    int mt = half * 4 + i;
                    acc[mt] = __builtin_amdgcn_wmma_f32_16x16x32_bf16(false, a4[i],
                                                                      false, bfr[ks],
                                                                      (short)0, acc[mt],
                                                                      false, false);
                }
            }
        }
        __syncthreads();   // done reading hsT before next chunk overwrites it
    }

    // combine: out[token] += g * (h@W2^T) + g * b2  (gate already folded into h)
    for (int mt = 0; mt < 8; ++mt) {
        #pragma unroll
        for (int r = 0; r < 8; ++r) {
            int dcol = wave * 128 + mt * 16 + r + 8 * hi;
            float v = acc[mt][r] + gn * b2e[dcol];
            atomicAdd(&out[(size_t)tokn * D_ + dcol], v);
        }
    }
}

// ---------------- host launch ----------------
extern "C" void kernel_launch(void* const* d_in, const int* in_sizes, int n_in,
                              void* d_out, int out_size, void* d_ws, size_t ws_size,
                              hipStream_t stream) {
    const float* x  = (const float*)d_in[0];
    const float* Wr = (const float*)d_in[1];
    const float* br = (const float*)d_in[2];
    const float* W1 = (const float*)d_in[3];
    const float* b1 = (const float*)d_in[4];
    const float* W2 = (const float*)d_in[5];
    const float* b2 = (const float*)d_in[6];
    float* out = (float*)d_out;

    // workspace carve (~170 KB total)
    char* p = (char*)d_ws;
    int*   counts = (int*)p;   p += 256;
    int*   pbase  = (int*)p;   p += 256;
    int*   a_e    = (int*)p;   p += (size_t)T_ * KTOP * 4;
    int*   a_slot = (int*)p;   p += (size_t)T_ * KTOP * 4;
    float* a_g    = (float*)p; p += (size_t)T_ * KTOP * 4;
    int*   perm   = (int*)p;   p += (size_t)(T_ * KTOP + E_ * 16) * 4;
    float* gslot  = (float*)p; p += (size_t)(T_ * KTOP + E_ * 16) * 4;
    int*   tileE  = (int*)p;   p += (size_t)MAX_TILES * 4;

    hipMemsetAsync(counts, 0, 8 * sizeof(int), stream);
    hipMemsetAsync(out, 0, (size_t)T_ * D_ * sizeof(float), stream);

    moe_router <<<T_ / 8, 256, 0, stream>>>(x, Wr, br, counts, a_e, a_slot, a_g);
    moe_prefix <<<1, 32, 0, stream>>>(counts, pbase, tileE, perm, gslot);
    moe_scatter<<<(T_ * KTOP + 255) / 256, 256, 0, stream>>>(a_e, a_slot, a_g, pbase, perm, gslot);
    moe_mlp    <<<MAX_TILES, 256, 0, stream>>>(x, W1, b1, W2, b2, perm, gslot, tileE, out);
}
